// BiLSTM_CRF_22454089023532
// MI455X (gfx1250) — matrix operand order repaired and checked by reading, not compiled, not run on hardware
//
#include <hip/hip_runtime.h>
#include <hip/hip_bf16.h>
#include <math.h>

typedef __bf16 bf16;
typedef __attribute__((ext_vector_type(16))) __bf16 v16bf;
typedef __attribute__((ext_vector_type(8)))  float  v8f;

#define B_   128
#define S_   512
#define E_   256
#define HD_  512
#define H_   256
#define T_   12
#define G4H  1024   // 4*H
#define NEGV -10000.0f

// ---------------------------------------------------------------------------
// WMMA fragment helpers (CDNA5 16x16x32 bf16 layouts, wave32)
// ---------------------------------------------------------------------------

// A-matrix 16x32 (MxK), row-major source with leading dim lda.
// lanes 0-15: M=lane, VGPR0-3 hold K=0..7, VGPR4-7 hold K=16..23
// lanes 16-31: M=lane-16, VGPR0-3 hold K=8..15, VGPR4-7 hold K=24..31
__device__ inline v16bf frag_a_rowmajor(const bf16* __restrict__ A, int lda,
                                        int m0, int k0) {
  int lane = threadIdx.x & 31;
  int l = lane & 15, hi = lane >> 4;
  const bf16* row = A + (size_t)(m0 + l) * lda + k0;
  v16bf a;
#pragma unroll
  for (int v = 0; v < 8; ++v) {
    int koff = (v < 4) ? (hi * 8 + 2 * v) : (16 + hi * 8 + 2 * (v - 4));
    a[2 * v]     = row[koff];
    a[2 * v + 1] = row[koff + 1];
  }
  return a;
}

// B-matrix 32x16 (KxN) taken from a weight stored (N,K) row-major, i.e.
// B[k][n] = W[n][k]. N=n0+lane%16; K order matches A: hi*8+2v (v<4), 16+hi*8+2(v-4).
__device__ inline v16bf frag_b_wT(const bf16* __restrict__ W, int ldw,
                                  int n0, int k0) {
  int lane = threadIdx.x & 31;
  int l = lane & 15, hi = lane >> 4;
  const bf16* col = W + (size_t)(n0 + l) * ldw + k0;
  v16bf b;
#pragma unroll
  for (int v = 0; v < 8; ++v) {
    int koff = (v < 4) ? (hi * 8 + 2 * v) : (16 + hi * 8 + 2 * (v - 4));
    b[2 * v]     = col[koff];
    b[2 * v + 1] = col[koff + 1];
  }
  return b;
}

__device__ inline v8f wmma_bf16(v16bf a, v16bf b, v8f c) {
  return __builtin_amdgcn_wmma_f32_16x16x32_bf16(
      false, a, false, b, (short)0, c, false, false);
}

__device__ inline float sigf(float x) { return 1.0f / (1.0f + __expf(-x)); }

// ---------------------------------------------------------------------------
// Prep kernels
// ---------------------------------------------------------------------------
__global__ void k_f32_to_bf16(const float* __restrict__ in,
                              bf16* __restrict__ out, int n) {
  int i = blockIdx.x * 256 + threadIdx.x;
  if (i < n) out[i] = (bf16)in[i];
}

__global__ void k_bias_add(const float* __restrict__ a,
                           const float* __restrict__ b,
                           float* __restrict__ o, int n) {
  int i = blockIdx.x * 256 + threadIdx.x;
  if (i < n) o[i] = a[i] + b[i];
}

// Pad W_tag (12 x 512) -> (16 x 512) bf16 with zero rows 12..15
__global__ void k_wtag_pad(const float* __restrict__ W, bf16* __restrict__ o) {
  int i = blockIdx.x * 256 + threadIdx.x;      // 16*512 = 8192
  if (i >= 16 * HD_) return;
  int n = i >> 9, k = i & 511;
  o[i] = (n < T_) ? (bf16)W[n * HD_ + k] : (bf16)0.0f;
}

// Embedding gather -> bf16, one block per (b,s) row
__global__ __launch_bounds__(256) void k_embed(const int* __restrict__ sent,
                                               const float* __restrict__ emb,
                                               bf16* __restrict__ xe) {
  int row = blockIdx.x;                  // b*S + s
  int tok = sent[row];
  xe[(size_t)row * E_ + threadIdx.x] = (bf16)emb[(size_t)tok * E_ + threadIdx.x];
}

// ---------------------------------------------------------------------------
// Input-projection GEMM: C[M,N] f32 = A[M,K] bf16 * W[N,K]^T bf16 + bias[N]
// block = 256 (8 waves). wave w: two m-tiles (32 rows), n-strip 64 wide.
// B fragments are shared by both m-tiles -> 2x reuse of the weight stream.
// ---------------------------------------------------------------------------
__global__ __launch_bounds__(256) void k_gemm_pre(const bf16* __restrict__ A,
                                                  const bf16* __restrict__ W,
                                                  const float* __restrict__ bias,
                                                  float* __restrict__ C,
                                                  int K, int N) {
  int wave = threadIdx.x >> 5;
  int m0 = (blockIdx.x * 8 + wave) * 32;
  int n0 = blockIdx.y * 64;
  v8f acc0[4] = {};
  v8f acc1[4] = {};
  for (int k0 = 0; k0 < K; k0 += 32) {
    if (k0 + 32 < K)
      __builtin_prefetch(A + (size_t)(m0 + (threadIdx.x & 15)) * K + k0 + 32, 0, 1);
    v16bf a0 = frag_a_rowmajor(A, K, m0, k0);
    v16bf a1 = frag_a_rowmajor(A, K, m0 + 16, k0);
#pragma unroll
    for (int i = 0; i < 4; ++i) {
      v16bf b = frag_b_wT(W, K, n0 + 16 * i, k0);
      acc0[i] = wmma_bf16(a0, b, acc0[i]);
      acc1[i] = wmma_bf16(a1, b, acc1[i]);
    }
  }
  int lane = threadIdx.x & 31, l = lane & 15, hi = lane >> 4;
#pragma unroll
  for (int i = 0; i < 4; ++i) {
    int col = n0 + 16 * i + l;
    float bs = bias[col];
#pragma unroll
    for (int v = 0; v < 8; ++v) {
      int row = m0 + hi * 8 + v;
      C[(size_t)row * N + col]        = acc0[i][v] + bs;
      C[(size_t)(row + 16) * N + col] = acc1[i][v] + bs;
    }
  }
}

// ---------------------------------------------------------------------------
// LSTM recurrence for one layer, both directions (blockIdx.y = dir).
// 8 blocks x 16 batch rows. h (bf16) and c (f32) resident in LDS.
// Per step:
//   1) issue GLOBAL_LOAD_ASYNC_TO_LDS_B128 copy of the 64KB pre(t) tile
//   2) z(16x1024) = h(16x256) @ Whh^T via WMMA (overlaps with async copy)
//   3) s_wait_asynccnt 0 + barrier, z += pre, gates, h/c update, write out
// ---------------------------------------------------------------------------
__global__ __launch_bounds__(256) void k_lstm(const float* __restrict__ preF,
                                              const float* __restrict__ preB,
                                              const bf16* __restrict__ WhhF,
                                              const bf16* __restrict__ WhhB,
                                              bf16* __restrict__ out) {
  __shared__ __align__(16) bf16  hbuf[16 * H_];      //  8 KB
  __shared__ __align__(16) float cbuf[16 * H_];      // 16 KB
  __shared__ __align__(16) float zbuf[16 * G4H];     // 64 KB
  __shared__ __align__(16) float pbuf[16 * G4H];     // 64 KB (pre tile)

  int dir   = blockIdx.y;
  int bbase = blockIdx.x * 16;
  const float* pre = dir ? preB : preF;
  const bf16*  Whh = dir ? WhhB : WhhF;

  for (int i = threadIdx.x; i < 16 * H_; i += 256) {
    hbuf[i] = (bf16)0.0f;
    cbuf[i] = 0.0f;
  }
  __syncthreads();

  int wave = threadIdx.x >> 5;
  int lane = threadIdx.x & 31, l = lane & 15, hi = lane >> 4;
  unsigned pbase = (unsigned)(size_t)(void*)pbuf;    // LDS byte offset (low 32b)

  for (int step = 0; step < S_; ++step) {
    int t = dir ? (S_ - 1 - step) : step;

    // ---- 1) async-stage pre(t) tile (16 x 1024 f32) into pbuf ----
    // 4096 b128 transfers; 16 per thread. Overlaps with the WMMA GEMM below.
#pragma unroll
    for (int it = 0; it < 16; ++it) {
      int idx = threadIdx.x + 256 * it;       // b128 index within tile
      int row = idx >> 8;                     // 256 b128 per 1024-float row
      int col = (idx & 255) * 4;              // float column
      const float* gp = pre + ((size_t)(bbase + row) * S_ + t) * G4H + col;
      unsigned ldsa = pbase + (unsigned)idx * 16u;
      asm volatile("global_load_async_to_lds_b128 %0, %1, off"
                   :: "v"(ldsa), "v"(gp) : "memory");
    }

    // ---- 2) z = h @ Whh^T : wave handles n-strip [wave*128, wave*128+128) ----
    v8f acc[8] = {};
    for (int k0 = 0; k0 < H_; k0 += 32) {
      v16bf a = frag_a_rowmajor(hbuf, H_, 0, k0);
#pragma unroll
      for (int i = 0; i < 8; ++i) {
        v16bf b = frag_b_wT(Whh, H_, wave * 128 + 16 * i, k0);
        acc[i] = wmma_bf16(a, b, acc[i]);
      }
    }

    // ---- 3) wait for the pre tile, combine, gates ----
    asm volatile("s_wait_asynccnt 0x0" ::: "memory");
    __syncthreads();

#pragma unroll
    for (int i = 0; i < 8; ++i) {
      int col = wave * 128 + 16 * i + l;
#pragma unroll
      for (int v = 0; v < 8; ++v) {
        int row = hi * 8 + v;
        zbuf[row * G4H + col] = acc[i][v] + pbuf[row * G4H + col];
      }
    }
    __syncthreads();

    for (int idx = threadIdx.x; idx < 16 * H_; idx += 256) {
      int bq = idx >> 8, j = idx & 255;
      const float* z = zbuf + bq * G4H;
      float ig = z[j];
      float fg = z[H_ + j];
      float gg = z[2 * H_ + j];
      float og = z[3 * H_ + j];
      float c = cbuf[idx];
      c = sigf(fg) * c + sigf(ig) * tanhf(gg);
      float h = sigf(og) * tanhf(c);
      cbuf[idx] = c;
      hbuf[idx] = (bf16)h;
      out[((size_t)(bbase + bq) * S_ + t) * HD_ + dir * H_ + j] = (bf16)h;
    }
    __syncthreads();
  }
}

// ---------------------------------------------------------------------------
// LayerNorm over HD=512, one block per row, bf16 in/out, f32 stats
// ---------------------------------------------------------------------------
__global__ __launch_bounds__(256) void k_ln(const bf16* __restrict__ X,
                                            const float* __restrict__ g,
                                            const float* __restrict__ bb,
                                            bf16* __restrict__ Y) {
  __shared__ float ls[8], lq[8];
  size_t row = blockIdx.x;
  const bf16* x = X + row * HD_;
  int j0 = threadIdx.x, j1 = threadIdx.x + 256;
  float v0 = (float)x[j0], v1 = (float)x[j1];
  float s = v0 + v1, q = v0 * v0 + v1 * v1;
#pragma unroll
  for (int o = 16; o >= 1; o >>= 1) {
    s += __shfl_down(s, o, 32);
    q += __shfl_down(q, o, 32);
  }
  int wave = threadIdx.x >> 5, lane = threadIdx.x & 31;
  if (lane == 0) { ls[wave] = s; lq[wave] = q; }
  __syncthreads();
  if (threadIdx.x == 0) {
    float ts = 0.f, tq = 0.f;
    for (int i = 0; i < 8; ++i) { ts += ls[i]; tq += lq[i]; }
    ls[0] = ts; lq[0] = tq;
  }
  __syncthreads();
  float mean = ls[0] * (1.0f / HD_);
  float var  = lq[0] * (1.0f / HD_) - mean * mean;
  float rstd = rsqrtf(var + 1e-5f);
  Y[row * HD_ + j0] = (bf16)((v0 - mean) * rstd * g[j0] + bb[j0]);
  Y[row * HD_ + j1] = (bf16)((v1 - mean) * rstd * g[j1] + bb[j1]);
}

// ---------------------------------------------------------------------------
// feats GEMM: (M x 512) bf16 * Wt(16 x 512)^T -> feats f32 (M x 12)
// one wave per 16-row m-tile (N padded to 16, rows 12..15 are zero)
// ---------------------------------------------------------------------------
__global__ __launch_bounds__(256) void k_feats(const bf16* __restrict__ X,
                                               const bf16* __restrict__ Wt,
                                               float* __restrict__ F) {
  int wave = threadIdx.x >> 5;
  int m0 = (blockIdx.x * 8 + wave) * 16;
  v8f acc = {};
  for (int k0 = 0; k0 < HD_; k0 += 32) {
    v16bf a = frag_a_rowmajor(X, HD_, m0, k0);
    v16bf b = frag_b_wT(Wt, HD_, 0, k0);
    acc = wmma_bf16(a, b, acc);
  }
  int lane = threadIdx.x & 31, l = lane & 15, hi = lane >> 4;
  if (l < T_) {
#pragma unroll
    for (int v = 0; v < 8; ++v) {
      int row = m0 + hi * 8 + v;
      F[(size_t)row * T_ + l] = acc[v];
    }
  }
}

// ---------------------------------------------------------------------------
// CRF forward: one wave per batch element, alpha in LDS
// ---------------------------------------------------------------------------
__global__ __launch_bounds__(32) void k_crf_fwd(const float* __restrict__ F,
                                                const float* __restrict__ trans,
                                                float* __restrict__ fwd) {
  __shared__ float alpha[T_], anew[T_], tr[T_ * T_];
  int b = blockIdx.x;
  int i = threadIdx.x;
  if (i < T_) alpha[i] = (i == 10) ? 0.0f : NEGV;
  for (int k = i; k < T_ * T_; k += 32) tr[k] = trans[k];
  __syncthreads();
  for (int t = 0; t < S_; ++t) {
    if (i < T_) {
      float m = -3.4e38f;
#pragma unroll
      for (int j = 0; j < T_; ++j) m = fmaxf(m, alpha[j] + tr[i * T_ + j]);
      float ssum = 0.0f;
#pragma unroll
      for (int j = 0; j < T_; ++j) ssum += __expf(alpha[j] + tr[i * T_ + j] - m);
      anew[i] = F[((size_t)b * S_ + t) * T_ + i] + m + __logf(ssum);
    }
    __syncthreads();
    if (i < T_) alpha[i] = anew[i];
    __syncthreads();
  }
  if (i == 0) {
    float m = -3.4e38f;
    for (int j = 0; j < T_; ++j) m = fmaxf(m, alpha[j] + tr[11 * T_ + j]);
    float ssum = 0.0f;
    for (int j = 0; j < T_; ++j) ssum += __expf(alpha[j] + tr[11 * T_ + j] - m);
    fwd[b] = m + __logf(ssum);
  }
}

// ---------------------------------------------------------------------------
// CRF gold score: one block per batch element
// ---------------------------------------------------------------------------
__global__ __launch_bounds__(64) void k_crf_gold(const float* __restrict__ F,
                                                 const int* __restrict__ tags,
                                                 const float* __restrict__ trans,
                                                 float* __restrict__ gold) {
  __shared__ float red[64];
  int b = blockIdx.x;
  const int* tg = tags + (size_t)b * S_;
  float acc = 0.0f;
  for (int s = threadIdx.x; s < S_; s += 64) {
    int cur = tg[s];
    int prev = (s == 0) ? 10 : tg[s - 1];
    acc += trans[cur * T_ + prev] + F[((size_t)b * S_ + s) * T_ + cur];
  }
  red[threadIdx.x] = acc;
  __syncthreads();
  for (int o = 32; o >= 1; o >>= 1) {
    if (threadIdx.x < o) red[threadIdx.x] += red[threadIdx.x + o];
    __syncthreads();
  }
  if (threadIdx.x == 0) gold[b] = red[0] + trans[11 * T_ + tg[S_ - 1]];
}

__global__ __launch_bounds__(128) void k_final(const float* __restrict__ fwd,
                                               const float* __restrict__ gold,
                                               float* __restrict__ out) {
  __shared__ float red[128];
  red[threadIdx.x] = fwd[threadIdx.x] - gold[threadIdx.x];
  __syncthreads();
  for (int o = 64; o >= 1; o >>= 1) {
    if (threadIdx.x < o) red[threadIdx.x] += red[threadIdx.x + o];
    __syncthreads();
  }
  if (threadIdx.x == 0) out[0] = red[0] / (float)B_;
}

// ---------------------------------------------------------------------------
// Host-side launch
// ---------------------------------------------------------------------------
extern "C" void kernel_launch(void* const* d_in, const int* in_sizes, int n_in,
                              void* d_out, int out_size, void* d_ws, size_t ws_size,
                              hipStream_t stream) {
  (void)in_sizes; (void)n_in; (void)out_size; (void)ws_size;

  // inputs in setup_inputs() dict (insertion) order
  const int*   sentence = (const int*)  d_in[0];
  const int*   tags     = (const int*)  d_in[1];
  /* seq_len = d_in[2] (all == S, unused) */
  const float* emb      = (const float*)d_in[3];
  const float* wih0f    = (const float*)d_in[4];
  const float* whh0f    = (const float*)d_in[5];
  const float* bih0f    = (const float*)d_in[6];
  const float* bhh0f    = (const float*)d_in[7];
  const float* wih0b    = (const float*)d_in[8];
  const float* whh0b    = (const float*)d_in[9];
  const float* bih0b    = (const float*)d_in[10];
  const float* bhh0b    = (const float*)d_in[11];
  const float* wih1f    = (const float*)d_in[12];
  const float* whh1f    = (const float*)d_in[13];
  const float* bih1f    = (const float*)d_in[14];
  const float* bhh1f    = (const float*)d_in[15];
  const float* wih1b    = (const float*)d_in[16];
  const float* whh1b    = (const float*)d_in[17];
  const float* bih1b    = (const float*)d_in[18];
  const float* bhh1b    = (const float*)d_in[19];
  const float* ln_g     = (const float*)d_in[20];
  const float* ln_b     = (const float*)d_in[21];
  const float* w_tag    = (const float*)d_in[22];
  const float* trans    = (const float*)d_in[23];

  const size_t MS = (size_t)B_ * S_;          // 65536 rows

  // workspace carve-out
  char* base = (char*)d_ws;
  size_t off = 0;
  auto carve = [&](size_t bytes) -> char* {
    char* p = base + off;
    off = (off + bytes + 255) & ~(size_t)255;
    return p;
  };
  bf16*  xe      = (bf16*) carve(MS * E_ * 2);
  bf16*  bwih0f  = (bf16*) carve((size_t)G4H * E_ * 2);
  bf16*  bwih0b  = (bf16*) carve((size_t)G4H * E_ * 2);
  bf16*  bwhh0f  = (bf16*) carve((size_t)G4H * H_ * 2);
  bf16*  bwhh0b  = (bf16*) carve((size_t)G4H * H_ * 2);
  bf16*  bwih1f  = (bf16*) carve((size_t)G4H * HD_ * 2);
  bf16*  bwih1b  = (bf16*) carve((size_t)G4H * HD_ * 2);
  bf16*  bwhh1f  = (bf16*) carve((size_t)G4H * H_ * 2);
  bf16*  bwhh1b  = (bf16*) carve((size_t)G4H * H_ * 2);
  float* bias0f  = (float*)carve(G4H * 4);
  float* bias0b  = (float*)carve(G4H * 4);
  float* bias1f  = (float*)carve(G4H * 4);
  float* bias1b  = (float*)carve(G4H * 4);
  bf16*  wtagp   = (bf16*) carve((size_t)16 * HD_ * 2);
  float* preF    = (float*)carve(MS * G4H * 4);     // 268 MB
  float* preB    = (float*)carve(MS * G4H * 4);     // 268 MB
  bf16*  x1      = (bf16*) carve(MS * HD_ * 2);
  bf16*  x2      = (bf16*) carve(MS * HD_ * 2);
  bf16*  xn      = (bf16*) carve(MS * HD_ * 2);
  float* feats   = (float*)carve(MS * T_ * 4);
  float* fwd     = (float*)carve(B_ * 4);
  float* gold    = (float*)carve(B_ * 4);

  // --- weight conversions to bf16 ---
  auto conv = [&](const float* src, bf16* dst, int n) {
    k_f32_to_bf16<<<(n + 255) / 256, 256, 0, stream>>>(src, dst, n);
  };
  conv(wih0f, bwih0f, G4H * E_);
  conv(wih0b, bwih0b, G4H * E_);
  conv(whh0f, bwhh0f, G4H * H_);
  conv(whh0b, bwhh0b, G4H * H_);
  conv(wih1f, bwih1f, G4H * HD_);
  conv(wih1b, bwih1b, G4H * HD_);
  conv(whh1f, bwhh1f, G4H * H_);
  conv(whh1b, bwhh1b, G4H * H_);
  k_bias_add<<<4, 256, 0, stream>>>(bih0f, bhh0f, bias0f, G4H);
  k_bias_add<<<4, 256, 0, stream>>>(bih0b, bhh0b, bias0b, G4H);
  k_bias_add<<<4, 256, 0, stream>>>(bih1f, bhh1f, bias1f, G4H);
  k_bias_add<<<4, 256, 0, stream>>>(bih1b, bhh1b, bias1b, G4H);
  k_wtag_pad<<<32, 256, 0, stream>>>(w_tag, wtagp);

  // --- embedding gather ---
  k_embed<<<(int)MS, 256, 0, stream>>>(sentence, emb, xe);

  dim3 gemmGrid((unsigned)(MS / 256), G4H / 64);

  // --- layer 0: input projections + recurrence ---
  k_gemm_pre<<<gemmGrid, 256, 0, stream>>>(xe, bwih0f, bias0f, preF, E_, G4H);
  k_gemm_pre<<<gemmGrid, 256, 0, stream>>>(xe, bwih0b, bias0b, preB, E_, G4H);
  k_lstm<<<dim3(B_ / 16, 2), 256, 0, stream>>>(preF, preB, bwhh0f, bwhh0b, x1);

  // --- layer 1: input projections + recurrence ---
  k_gemm_pre<<<gemmGrid, 256, 0, stream>>>(x1, bwih1f, bias1f, preF, HD_, G4H);
  k_gemm_pre<<<gemmGrid, 256, 0, stream>>>(x1, bwih1b, bias1b, preB, HD_, G4H);
  k_lstm<<<dim3(B_ / 16, 2), 256, 0, stream>>>(preF, preB, bwhh1f, bwhh1b, x2);

  // --- layernorm + tag projection ---
  k_ln<<<(int)MS, 256, 0, stream>>>(x2, ln_g, ln_b, xn);
  k_feats<<<(int)(MS / 128), 256, 0, stream>>>(xn, wtagp, feats);

  // --- CRF ---
  k_crf_fwd<<<B_, 32, 0, stream>>>(feats, trans, fwd);
  k_crf_gold<<<B_, 64, 0, stream>>>(feats, tags, trans, gold);
  k_final<<<1, 128, 0, stream>>>(fwd, gold, (float*)d_out);
}
